// Feedback_Queries_75187697483912
// MI455X (gfx1250) — matrix operand order repaired and checked by reading, not compiled, not run on hardware
//
#include <hip/hip_runtime.h>
#include <cstdint>
#include <cstddef>

typedef __bf16 bf16;
typedef __attribute__((ext_vector_type(8)))  __bf16 v8bf;
typedef __attribute__((ext_vector_type(16))) __bf16 v16bf;
typedef __attribute__((ext_vector_type(8)))  float  v8f;
typedef __attribute__((ext_vector_type(4)))  float  v4f;
typedef __attribute__((ext_vector_type(4)))  uint32_t u32x4;

#define THREADS 256
#define WAVES 8

// ---- model constants ----
constexpr int NF    = 101;
constexpr int UNITS = 125;
constexpr int G3    = 375;    // 3*UNITS
constexpr int NPAD  = 384;    // padded gate width (24 tiles of 16)
constexpr int KPAD  = 128;    // padded K (4 k-tiles of 32) for both 101 and 125
constexpr int OUTC  = 202;    // 2*NF
constexpr int OPAD  = 208;    // 13 tiles of 16
constexpr int KS    = 25250;  // UNITS*OUTC
constexpr int NW    = 25452;  // KS + OUTC
constexpr int TIN   = 56;     // 70 - LAG
constexpr int TPRED = 28;     // GAMMA
constexpr int BROWS = 16;     // batch rows per workgroup
constexpr int BATCH = 4096;
constexpr float CC  = 0.5413248546129181f;  // log(expm1(1))

// ---- LDS layout (bytes) ----
constexpr int OFF_WK   = 0;                          // 384*128 bf16 = 96KB
constexpr int OFF_U    = OFF_WK  + NPAD*KPAD*2;      // 96KB
constexpr int OFF_WV   = OFF_U   + NPAD*KPAD*2;      // 208*128 bf16 = 52KB
constexpr int OFF_XA   = OFF_WV  + OPAD*KPAD*2;      // 16*128 bf16 A-tile
constexpr int OFF_HA   = OFF_XA  + BROWS*KPAD*2;     // 16*128 bf16 A-tile
constexpr int OFF_HF   = OFF_HA  + BROWS*KPAD*2;     // 16*128 f32 (master h)
constexpr int OFF_SSUM = OFF_HF  + BROWS*KPAD*4;     // [384][16] f32 n-major stage (aliased y [208][16])
constexpr int OFF_MIH  = OFF_SSUM + NPAD*BROWS*4;    // [144][16] f32 (mi, h-gate cols 240..383)
constexpr int OFF_BSUM = OFF_MIH + 144*BROWS*4;      // 384 f32 (b0+b1)
constexpr int OFF_B1   = OFF_BSUM + NPAD*4;          // 384 f32 (b1)
constexpr int OFF_BV   = OFF_B1  + NPAD*4;           // 208 f32 (variational bias)
constexpr int LDS_BYTES = OFF_BV + OPAD*4;           // = 303936 < 320KB
static_assert(LDS_BYTES == 303936, "lds layout");

// ---- workspace layout (bytes) ----
constexpr size_t WS_WV   = 0;                                  // 28*208*128 bf16
constexpr size_t WS_BV   = WS_WV + (size_t)TPRED*OPAD*KPAD*2;  // 28*208 f32
constexpr size_t WS_WKT  = WS_BV + (size_t)TPRED*OPAD*4;       // 384*128 bf16
constexpr size_t WS_UT   = WS_WKT + (size_t)NPAD*KPAD*2;       // 384*128 bf16
constexpr size_t WS_END  = WS_UT + (size_t)NPAD*KPAD*2;        // ~1.71 MB

// ---- scalar helpers ----
__device__ __forceinline__ float softplus_f(float x) {
  return (x > 15.0f) ? x : log1pf(__expf(x));
}
__device__ __forceinline__ float sigmoid_f(float x) {
  return 1.0f / (1.0f + __expf(-x));
}
__device__ __forceinline__ float tanh_f(float x) {
  return 1.0f - 2.0f / (__expf(2.0f * x) + 1.0f);
}

// ---- WMMA fragment loads from LDS ----
// A (16x32 bf16, row-major tile [16][128]): lane L, M=L&15;
//   L<16 : K = 0..7 (VGPR0-3) and 16..23 (VGPR4-7)
//   L>=16: K = 8..15           and 24..31
__device__ __forceinline__ v16bf load_a_frag(const bf16* base, int kt, int lane) {
  const bf16* p = base + ((lane & 15) << 7) + (kt << 5) + ((lane >> 4) << 3);
  v8bf lo = *(const v8bf*)(p);
  v8bf hi = *(const v8bf*)(p + 16);
  return __builtin_shufflevector(lo, hi, 0,1,2,3,4,5,6,7,8,9,10,11,12,13,14,15);
}
// B (32x16 bf16, column-major tile [n][128]): lane L, N=tile*16+(L&15);
//   L<16 : K = 0..7 and 16..23; L>=16: K = 8..15 and 24..31
__device__ __forceinline__ v16bf load_b_frag(const bf16* base, int nt, int kt, int lane) {
  const bf16* p = base + (((nt << 4) + (lane & 15)) << 7) + (kt << 5) + ((lane >> 4) << 3);
  v8bf lo = *(const v8bf*)(p);
  v8bf hi = *(const v8bf*)(p + 16);
  return __builtin_shufflevector(lo, hi, 0,1,2,3,4,5,6,7,8,9,10,11,12,13,14,15);
}

// store a lane's 8 accumulator elements (+bias) into an n-major [n][16] stage
__device__ __forceinline__ void stage_store(float* stage, int col, int mr,
                                            const v8f& acc, float bias) {
  v4f lo = {acc[0] + bias, acc[1] + bias, acc[2] + bias, acc[3] + bias};
  v4f hi = {acc[4] + bias, acc[5] + bias, acc[6] + bias, acc[7] + bias};
  *(v4f*)(&stage[col * BROWS + mr])     = lo;
  *(v4f*)(&stage[col * BROWS + mr + 4]) = hi;
}

// ---- one GRU step for the workgroup's 16 rows ----
__device__ __forceinline__ void gru_step(const bf16* sWk, const bf16* sU,
                                         const bf16* sXa, bf16* sHa, float* sHf,
                                         float* sSum, float* sMih,
                                         const float* sBsum, const float* sB1,
                                         int tid, int lane) {
  const int wave = tid >> 5;
  // hoist A fragments: 4 k-tiles each of x and h, reused across 3 N-tiles
  v16bf ax[4], ah[4];
  #pragma unroll
  for (int kt = 0; kt < 4; ++kt) {
    ax[kt] = load_a_frag(sXa, kt, lane);
    ah[kt] = load_a_frag(sHa, kt, lane);
  }
  #pragma unroll
  for (int j = 0; j < 3; ++j) {
    const int nt = wave * 3 + j;
    v8f accX = {0.f,0.f,0.f,0.f,0.f,0.f,0.f,0.f};
    v8f accH = {0.f,0.f,0.f,0.f,0.f,0.f,0.f,0.f};
    #pragma unroll
    for (int kt = 0; kt < 4; ++kt) {
      v16bf bm = load_b_frag(sWk, nt, kt, lane);
      accX = __builtin_amdgcn_wmma_f32_16x16x32_bf16(false, ax[kt], false, bm,
                                                     (short)0, accX, false, false);
    }
    #pragma unroll
    for (int kt = 0; kt < 4; ++kt) {
      v16bf bm = load_b_frag(sU, nt, kt, lane);
      accH = __builtin_amdgcn_wmma_f32_16x16x32_bf16(false, ah[kt], false, bm,
                                                     (short)0, accH, false, false);
    }
    const int n  = (nt << 4) + (lane & 15);
    const int mr = (lane >> 4) << 3;
    v8f s;
    #pragma unroll
    for (int i = 0; i < 8; ++i) s[i] = accX[i] + accH[i];
    stage_store(sSum, n, mr, s, sBsum[n]);
    if (nt >= 15)  // columns 240..383 hold the h-gate chunk (>=250 used)
      stage_store(sMih, n - 240, mr, accH, sB1[n]);
  }
  __syncthreads();
  // elementwise gate combine: 16*125 items, j-major (conflict-free LDS reads)
  for (int e = tid; e < BROWS * UNITS; e += THREADS) {
    const int j = e >> 4, m = e & 15;
    const float sz = sSum[j * BROWS + m];
    const float sr = sSum[(j + UNITS) * BROWS + m];
    const float sh = sSum[(j + 2 * UNITS) * BROWS + m];
    const float rh = sMih[(j + 10) * BROWS + m];  // (j+250)-240 ; == mi_h + b1
    const float xh = sh - rh;                      // mx_h + b0
    const float z  = sigmoid_f(sz);
    const float r  = sigmoid_f(sr);
    const float hh = tanh_f(xh + r * rh);
    const float hp = sHf[m * KPAD + j];
    const float hn = z * hp + (1.0f - z) * hh;
    sHf[m * KPAD + j] = hn;
    sHa[m * KPAD + j] = (bf16)hn;
  }
  __syncthreads();
}

// ---- variational dense: y = h @ Wv + bias (y staged n-major [208][16]) ----
__device__ __forceinline__ void dense_step(const bf16* sWv, const bf16* sHa,
                                           const float* sBv, float* ystage,
                                           int tid, int lane) {
  const int wave = tid >> 5;
  v16bf ah[4];
  #pragma unroll
  for (int kt = 0; kt < 4; ++kt) ah[kt] = load_a_frag(sHa, kt, lane);
  for (int nt = wave; nt < 13; nt += WAVES) {
    v8f acc = {0.f,0.f,0.f,0.f,0.f,0.f,0.f,0.f};
    #pragma unroll
    for (int kt = 0; kt < 4; ++kt) {
      v16bf bm = load_b_frag(sWv, nt, kt, lane);
      acc = __builtin_amdgcn_wmma_f32_16x16x32_bf16(false, ah[kt], false, bm,
                                                    (short)0, acc, false, false);
    }
    const int n  = (nt << 4) + (lane & 15);
    const int mr = (lane >> 4) << 3;
    stage_store(ystage, n, mr, acc, sBv[n]);
  }
  __syncthreads();
}

// ---- prep: transpose+pad W_k and U into bf16 B-layout ----
__global__ void prep_wk_u_kernel(const float* __restrict__ Wk, const float* __restrict__ U,
                                 bf16* __restrict__ wkt, bf16* __restrict__ ut) {
  const int idx = blockIdx.x * THREADS + threadIdx.x;
  if (idx >= 2 * NPAD * KPAD) return;
  const int which = idx / (NPAD * KPAD);
  const int r = idx - which * (NPAD * KPAD);
  const int n = r >> 7, k = r & 127;
  if (which == 0)
    wkt[r] = (k < NF && n < G3) ? (bf16)Wk[k * G3 + n] : (bf16)0.0f;
  else
    ut[r]  = (k < UNITS && n < G3) ? (bf16)U[k * G3 + n] : (bf16)0.0f;
}

// ---- prep: build per-step variational kernels (28 x [208][128] bf16) + biases ----
__global__ void prep_wv_kernel(const float* __restrict__ dv_loc, const float* __restrict__ dv_rho,
                               const float* __restrict__ eps_w0, const float* __restrict__ eps_w,
                               bf16* __restrict__ wv, float* __restrict__ biasg) {
  const int idx = blockIdx.x * THREADS + threadIdx.x;
  const int per = OPAD * KPAD;  // 26624
  if (idx < TPRED * per) {
    const int s = idx / per, r = idx - s * per;
    const int n = r >> 7, k = r & 127;
    float v = 0.0f;
    if (k < UNITS && n < OUTC) {
      const int i = k * OUTC + n;
      const float eps = (s == 0) ? eps_w0[i] : eps_w[(s - 1) * NW + i];
      v = dv_loc[i] + (1e-5f + 0.01f * softplus_f(CC + dv_rho[i])) * eps;
    }
    wv[idx] = (bf16)v;
  } else {
    const int j = idx - TPRED * per;
    if (j < TPRED * OPAD) {
      const int s = j / OPAD, n = j - s * OPAD;
      float v = 0.0f;
      if (n < OUTC) {
        const int i = KS + n;
        const float eps = (s == 0) ? eps_w0[i] : eps_w[(s - 1) * NW + i];
        v = dv_loc[i] + (1e-5f + 0.01f * softplus_f(CC + dv_rho[i])) * eps;
      }
      biasg[j] = v;
    }
  }
}

// ---- persistent main kernel: 256 WGs x 16 rows, full 83-step recurrence on-chip ----
__global__ __launch_bounds__(THREADS, 1)
void gru_forecast_kernel(const float* __restrict__ inputs,
                         const float* __restrict__ bgru,
                         const float* __restrict__ eps_s,
                         const bf16* __restrict__ wkt,
                         const bf16* __restrict__ ut,
                         const bf16* __restrict__ wvg,
                         const float* __restrict__ biasg,
                         float* __restrict__ out) {
  extern __shared__ char smem[];
  bf16*  sWk   = (bf16*)(smem + OFF_WK);
  bf16*  sU    = (bf16*)(smem + OFF_U);
  bf16*  sWv   = (bf16*)(smem + OFF_WV);
  bf16*  sXa   = (bf16*)(smem + OFF_XA);
  bf16*  sHa   = (bf16*)(smem + OFF_HA);
  float* sHf   = (float*)(smem + OFF_HF);
  float* sSum  = (float*)(smem + OFF_SSUM);
  float* sMih  = (float*)(smem + OFF_MIH);
  float* sBsum = (float*)(smem + OFF_BSUM);
  float* sB1   = (float*)(smem + OFF_B1);
  float* sBv   = (float*)(smem + OFF_BV);

  const int tid  = threadIdx.x;
  const int lane = tid & 31;
  const int m0   = blockIdx.x * BROWS;

  // setup: weights -> LDS (b128 copies), biases, zero h
  {
    const u32x4* s1 = (const u32x4*)wkt;
    const u32x4* s2 = (const u32x4*)ut;
    u32x4* d1 = (u32x4*)sWk;
    u32x4* d2 = (u32x4*)sU;
    for (int e = tid; e < NPAD * KPAD / 8; e += THREADS) { d1[e] = s1[e]; d2[e] = s2[e]; }
    for (int n = tid; n < NPAD; n += THREADS) {
      const float b0v = (n < G3) ? bgru[n] : 0.0f;
      const float b1v = (n < G3) ? bgru[G3 + n] : 0.0f;
      sBsum[n] = b0v + b1v;
      sB1[n]   = b1v;
    }
    for (int e = tid; e < BROWS * KPAD; e += THREADS) { sHf[e] = 0.0f; sHa[e] = (bf16)0.0f; }
  }
  __syncthreads();

  // warm-up phase: 56 GRU steps over the observed inputs
  for (int t = 0; t < TIN; ++t) {
    for (int e = tid; e < BROWS * KPAD; e += THREADS) {
      const int m = e >> 7, f = e & 127;
      float v = 0.0f;
      if (f < NF) v = inputs[((size_t)(m0 + m) * 70 + t) * NF + f];
      sXa[e] = (bf16)v;
    }
    if (t + 1 < TIN && tid < BROWS)
      __builtin_prefetch(&inputs[((size_t)(m0 + tid) * 70 + (t + 1)) * NF], 0, 0);
    __syncthreads();
    gru_step(sWk, sU, sXa, sHa, sHf, sSum, sMih, sBsum, sB1, tid, lane);
  }

  // autoregressive phase: 28 dense projections interleaved with 27 GRU steps
  for (int t = 0; t < TPRED; ++t) {
    if (t > 0)
      gru_step(sWk, sU, sXa, sHa, sHf, sSum, sMih, sBsum, sB1, tid, lane);
    {
      const u32x4* sv = (const u32x4*)(wvg + (size_t)t * OPAD * KPAD);
      u32x4* dv = (u32x4*)sWv;
      for (int e = tid; e < OPAD * KPAD / 8; e += THREADS) dv[e] = sv[e];
      for (int n = tid; n < OPAD; n += THREADS) sBv[n] = biasg[t * OPAD + n];
    }
    __syncthreads();
    dense_step(sWv, sHa, sBv, sSum, tid, lane);  // y staged n-major [208][16]
    // epilogue: emit (loc, scale) coalesced, build next x from eps_s
    for (int e = tid; e < BROWS * NF; e += THREADS) {
      const int m = e / NF, j = e - m * NF;
      const float loc = sSum[j * BROWS + m];
      const float raw = sSum[(j + NF) * BROWS + m];
      const float sc  = 1e-5f + 0.05f * softplus_f(CC + raw);
      const size_t ob = ((size_t)(m0 + m) * TPRED + t) * OUTC;
      out[ob + j]      = loc;
      out[ob + NF + j] = sc;
      if (t < TPRED - 1) {
        const float es = eps_s[((size_t)t * BATCH + (m0 + m)) * NF + j];
        sXa[m * KPAD + j] = (bf16)(loc + sc * es);  // pad cols stay zero
      }
    }
    __syncthreads();
  }
}

extern "C" void kernel_launch(void* const* d_in, const int* in_sizes, int n_in,
                              void* d_out, int out_size, void* d_ws, size_t ws_size,
                              hipStream_t stream) {
  (void)in_sizes; (void)n_in; (void)out_size; (void)ws_size;
  const float* inputs = (const float*)d_in[0];
  const float* W_k    = (const float*)d_in[1];
  const float* U      = (const float*)d_in[2];
  const float* b      = (const float*)d_in[3];
  const float* dv_loc = (const float*)d_in[4];
  const float* dv_rho = (const float*)d_in[5];
  const float* eps_w0 = (const float*)d_in[6];
  const float* eps_w  = (const float*)d_in[7];
  const float* eps_s  = (const float*)d_in[8];
  float* out = (float*)d_out;

  char* ws = (char*)d_ws;
  bf16*  wvg   = (bf16*)(ws + WS_WV);
  float* biasg = (float*)(ws + WS_BV);
  bf16*  wkt   = (bf16*)(ws + WS_WKT);
  bf16*  ut    = (bf16*)(ws + WS_UT);

  {
    const int total = 2 * NPAD * KPAD;
    prep_wk_u_kernel<<<(total + THREADS - 1) / THREADS, THREADS, 0, stream>>>(W_k, U, wkt, ut);
  }
  {
    const int total = TPRED * OPAD * KPAD + TPRED * OPAD;
    prep_wv_kernel<<<(total + THREADS - 1) / THREADS, THREADS, 0, stream>>>(
        dv_loc, dv_rho, eps_w0, eps_w, wvg, biasg);
  }

  hipFuncSetAttribute((const void*)gru_forecast_kernel,
                      hipFuncAttributeMaxDynamicSharedMemorySize, LDS_BYTES);
  gru_forecast_kernel<<<BATCH / BROWS, THREADS, LDS_BYTES, stream>>>(
      inputs, b, eps_s, wkt, ut, wvg, biasg, out);
}
